// BaselineModel_35175782154746
// MI455X (gfx1250) — matrix operand order repaired and checked by reading, not compiled, not run on hardware
//
#include <hip/hip_runtime.h>

// ---------------------------------------------------------------------------
// CDNA5 (gfx1250) bf16 WMMA pipeline for SiLU-attention block.
// B=4, S=1024, H=2048, NH=16, HD=128, MAXLEN=1024.
// ---------------------------------------------------------------------------

#define B_  4
#define S_  1024
#define H_  2048
#define NH_ 16
#define HD_ 128

typedef __attribute__((ext_vector_type(16))) __bf16 v16bf;
typedef __attribute__((ext_vector_type(8)))  float  v8f;

union BF16Frag {
    v16bf v;
    unsigned int   u[8];
    unsigned short s[16];
};

__device__ __forceinline__ unsigned short f2bf(float f) {
    unsigned int u = __float_as_uint(f);
    unsigned int r = u + 0x7fffu + ((u >> 16) & 1u);   // round-to-nearest-even
    return (unsigned short)(r >> 16);
}
__device__ __forceinline__ float bf2f(unsigned short s) {
    return __uint_as_float(((unsigned int)s) << 16);
}
__device__ __forceinline__ v8f wmma_bf16(const v16bf& a, const v16bf& b, v8f c) {
    return __builtin_amdgcn_wmma_f32_16x16x32_bf16(false, a, false, b, (short)0, c, false, false);
}
__device__ __forceinline__ float silu(float x) {
    return x / (1.0f + __expf(-x));
}
__device__ __forceinline__ unsigned long long pack4bf(float x, float y, float z, float w) {
    return (unsigned long long)f2bf(x)
         | ((unsigned long long)f2bf(y) << 16)
         | ((unsigned long long)f2bf(z) << 32)
         | ((unsigned long long)f2bf(w) << 48);
}

// ---------------------------------------------------------------------------
// fp32 -> bf16 conversion (vectorized x4)
// ---------------------------------------------------------------------------
__global__ __launch_bounds__(256) void cvt_f32_to_bf16(const float* __restrict__ src,
                                                       unsigned short* __restrict__ dst,
                                                       int n4) {
    int i = blockIdx.x * blockDim.x + threadIdx.x;
    if (i < n4) {
        float4 f = *(const float4*)(src + (size_t)i * 4);
        *(unsigned long long*)(dst + (size_t)i * 4) = pack4bf(f.x, f.y, f.z, f.w);
    }
}

// ---------------------------------------------------------------------------
// fp32 [K,N] -> bf16 [N,K] transpose-convert (tiled via LDS), K=N=2048 here.
// ---------------------------------------------------------------------------
__global__ __launch_bounds__(256) void cvtT_f32_to_bf16(const float* __restrict__ src,
                                                        unsigned short* __restrict__ dst,
                                                        int Kd, int Nd) {
    constexpr int TST = 36;   // 32 + pad, keeps 8B alignment
    __shared__ unsigned short tile[32 * TST];
    const int n0 = blockIdx.x * 32, k0 = blockIdx.y * 32;
    const int r  = threadIdx.x >> 3;
    const int c4 = (threadIdx.x & 7) * 4;
    {
        const float4 f = *(const float4*)(src + (size_t)(k0 + r) * Nd + n0 + c4);
        *(unsigned long long*)&tile[r * TST + c4] = pack4bf(f.x, f.y, f.z, f.w);
    }
    __syncthreads();
    {
        unsigned long long p = (unsigned long long)tile[(c4 + 0) * TST + r]
                             | ((unsigned long long)tile[(c4 + 1) * TST + r] << 16)
                             | ((unsigned long long)tile[(c4 + 2) * TST + r] << 32)
                             | ((unsigned long long)tile[(c4 + 3) * TST + r] << 48);
        *(unsigned long long*)(dst + (size_t)(n0 + r) * Kd + k0 + c4) = p;
    }
}

// ---------------------------------------------------------------------------
// bf16 V [B,S,H] -> VT [B,NH,HD,S] transpose (per-head), tiled via LDS.
// ---------------------------------------------------------------------------
__global__ __launch_bounds__(256) void transpose_v_bf16(const unsigned short* __restrict__ Vp,
                                                        unsigned short* __restrict__ VT) {
    constexpr int TST = 36;
    __shared__ unsigned short tile[32 * TST];
    const int s0 = blockIdx.x * 32, d0 = blockIdx.y * 32;
    const int bh = blockIdx.z;                 // b*NH + h
    const int b  = bh >> 4, h = bh & 15;
    const int r  = threadIdx.x >> 3;
    const int c4 = (threadIdx.x & 7) * 4;
    {
        const unsigned short* g = Vp + (size_t)(b * S_ + s0 + r) * H_ + h * HD_ + d0 + c4;
        *(unsigned long long*)&tile[r * TST + c4] = *(const unsigned long long*)g;
    }
    __syncthreads();
    {
        unsigned long long p = (unsigned long long)tile[(c4 + 0) * TST + r]
                             | ((unsigned long long)tile[(c4 + 1) * TST + r] << 16)
                             | ((unsigned long long)tile[(c4 + 2) * TST + r] << 32)
                             | ((unsigned long long)tile[(c4 + 3) * TST + r] << 48);
        *(unsigned long long*)(VT + ((size_t)bh * HD_ + d0 + r) * S_ + s0 + c4) = p;
    }
}

// ---------------------------------------------------------------------------
// Tiled bf16 WMMA GEMM:  out = act(A[M,K] @ W[K,N] + bias[N]),
// with W pre-transposed as WT[N,K]. Block tile 128x256x32, 8 waves (2x4),
// wave tile 64x64 = 16 WMMA per k-step. Register double-buffered global loads.
// ---------------------------------------------------------------------------
template<bool SILU, bool OUTF32>
__global__ __launch_bounds__(256) void gemm_bias_act(
        const unsigned short* __restrict__ A,    // [M,K] bf16
        const unsigned short* __restrict__ WT,   // [N,K] bf16 (pre-transposed)
        const float* __restrict__ bias,          // [N]
        unsigned short* __restrict__ obf,        // bf16 out (if !OUTF32)
        float* __restrict__ of32,                // f32  out (if OUTF32)
        int M, int N, int K)
{
    constexpr int BM = 128, BN = 256, BK = 32;
    constexpr int AST = BK + 8;   // 40
    constexpr int BST = BK + 8;   // 40
    __shared__ unsigned short As [BM * AST];
    __shared__ unsigned short Bst[BN * BST];

    const int tid   = threadIdx.x;
    const int wave  = tid >> 5;
    const int lane  = tid & 31;
    const int lm    = lane & 15;
    const int lhalf = lane >> 4;
    const int m0 = blockIdx.y * BM;
    const int n0 = blockIdx.x * BN;
    const int wr = wave >> 2;     // 0..1 -> 64-row band
    const int wc = wave & 3;      // 0..3 -> 64-col band

    const v8f vzero = {0.f, 0.f, 0.f, 0.f, 0.f, 0.f, 0.f, 0.f};
    v8f acc[4][4];
#pragma unroll
    for (int i = 0; i < 4; i++)
#pragma unroll
        for (int j = 0; j < 4; j++) acc[i][j] = vzero;

    // per-thread staging addresses
    const int arow = tid >> 1, acg = (tid & 1) * 16;
    const unsigned short* aptr = A + (size_t)(m0 + arow) * K + acg;
    const int brow = tid;   // one 32-wide row of WT per thread
    const unsigned short* bptr = WT + (size_t)(n0 + brow) * K;

    uint4 areg[2], breg[4];
    areg[0] = *(const uint4*)(aptr);
    areg[1] = *(const uint4*)(aptr + 8);
#pragma unroll
    for (int i = 0; i < 4; i++) breg[i] = *(const uint4*)(bptr + i * 8);

    for (int k0 = 0;;) {
        // --- commit staged registers to LDS ---
        *(uint4*)&As[arow * AST + acg]     = areg[0];
        *(uint4*)&As[arow * AST + acg + 8] = areg[1];
#pragma unroll
        for (int i = 0; i < 4; i++) *(uint4*)&Bst[brow * BST + i * 8] = breg[i];
        __syncthreads();

        const int k1 = k0 + BK;
        if (k1 < K) {   // prefetch next tile into registers (overlaps WMMA burst)
            areg[0] = *(const uint4*)(aptr + k1);
            areg[1] = *(const uint4*)(aptr + k1 + 8);
#pragma unroll
            for (int i = 0; i < 4; i++) breg[i] = *(const uint4*)(bptr + k1 + i * 8);
        }

        // --- fragments + 16 WMMA ---
        BF16Frag af[4], bfr[4];
#pragma unroll
        for (int i = 0; i < 4; i++) {
            const unsigned short* p = &As[(64 * wr + 16 * i + lm) * AST];
#pragma unroll
            for (int v = 0; v < 4; v++)
                af[i].u[v] = *(const unsigned int*)&p[8 * lhalf + 2 * v];
#pragma unroll
            for (int v = 0; v < 4; v++)
                af[i].u[4 + v] = *(const unsigned int*)&p[16 + 8 * lhalf + 2 * v];
        }
#pragma unroll
        for (int j = 0; j < 4; j++) {
            const unsigned short* p = &Bst[(64 * wc + 16 * j + lm) * BST];
#pragma unroll
            for (int v = 0; v < 4; v++)
                bfr[j].u[v] = *(const unsigned int*)&p[8 * lhalf + 2 * v];
#pragma unroll
            for (int v = 0; v < 4; v++)
                bfr[j].u[4 + v] = *(const unsigned int*)&p[16 + 8 * lhalf + 2 * v];
        }
#pragma unroll
        for (int i = 0; i < 4; i++)
#pragma unroll
            for (int j = 0; j < 4; j++)
                acc[i][j] = wmma_bf16(af[i].v, bfr[j].v, acc[i][j]);
        __syncthreads();
        k0 = k1;
        if (k0 >= K) break;
    }

    // --- epilogue: bias (+ SiLU), store ---
#pragma unroll
    for (int i = 0; i < 4; i++) {
#pragma unroll
        for (int j = 0; j < 4; j++) {
            int ncol = n0 + 64 * wc + 16 * j + lm;
            float bv = bias[ncol];
#pragma unroll
            for (int v = 0; v < 8; v++) {
                int mrow = m0 + 64 * wr + 16 * i + v + 8 * lhalf;
                float x = acc[i][j][v] + bv;
                if (SILU) x = silu(x);
                if (OUTF32) of32[(size_t)mrow * N + ncol] = x;
                else        obf [(size_t)mrow * N + ncol] = f2bf(x);
            }
        }
    }
}

// ---------------------------------------------------------------------------
// Attention: scores = silu(scale*QK^T + rel_bias) causal-masked, out = attn@V,
// gated = out * U. One block = (b, h, 128 q rows); wave = 16 q rows.
// V consumed via pre-transposed VT[B,NH,HD,S].
// ---------------------------------------------------------------------------
__global__ __launch_bounds__(256) void attn_silu_kernel(
        const unsigned short* __restrict__ Qp,   // [B,S,H] bf16 (head-interleaved)
        const unsigned short* __restrict__ Kp,   // [B,S,H] bf16
        const unsigned short* __restrict__ VT,   // [B,NH,HD,S] bf16
        const unsigned short* __restrict__ Up,   // [B,S,H] bf16
        const float* __restrict__ rel_table,     // [2*S-1, NH]
        unsigned short* __restrict__ gated)      // [B,S,H] bf16
{
    constexpr int KST = HD_ + 8;  // 136: K chunk row stride
    constexpr int VST = 40;       // Vst[hd][key] stride (32 keys + pad)
    constexpr int SST = 40;       // attn staging row stride
    __shared__ unsigned short Ks  [32 * KST];
    __shared__ unsigned short Vst [HD_ * VST];
    __shared__ unsigned short Attn[8 * 16 * SST];

    const int qb = blockIdx.x, h = blockIdx.y, b = blockIdx.z;
    const int bh = b * NH_ + h;
    const int q0 = qb * 128;
    const int tid   = threadIdx.x;
    const int wave  = tid >> 5;
    const int lane  = tid & 31;
    const int lm    = lane & 15;
    const int lhalf = lane >> 4;
    const int qw = q0 + 16 * wave;              // wave's first q row
    const float scale = 0.08838834764831845f;   // 1/sqrt(128)

    // Q fragments for this wave's 16 rows: 4 chunks of 32 along HD
    BF16Frag qf[4];
    {
        const unsigned short* qrow = Qp + ((size_t)(b * S_ + qw + lm) * H_ + h * HD_);
#pragma unroll
        for (int c = 0; c < 4; c++) {
            const unsigned short* p = qrow + c * 32;
#pragma unroll
            for (int v = 0; v < 4; v++)
                qf[c].u[v] = *(const unsigned int*)&p[8 * lhalf + 2 * v];
#pragma unroll
            for (int v = 0; v < 4; v++)
                qf[c].u[4 + v] = *(const unsigned int*)&p[16 + 8 * lhalf + 2 * v];
        }
    }

    const v8f vzero = {0.f, 0.f, 0.f, 0.f, 0.f, 0.f, 0.f, 0.f};
    v8f oacc[8];
#pragma unroll
    for (int j = 0; j < 8; j++) oacc[j] = vzero;

    const int kmax_blk = q0 + 127;
    const int kmax_w   = qw + 15;

    // per-thread staging addresses
    const int keyl = tid >> 3, kcb = (tid & 7) * 16;           // K: 32 keys x 128 hd
    const unsigned short* kptr = Kp + ((size_t)(b * S_ + keyl) * H_ + h * HD_ + kcb);
    const int vhd = tid >> 1, vcg = (tid & 1) * 16;            // V: 128 hd x 32 keys
    const unsigned short* vptr = VT + ((size_t)bh * HD_ + vhd) * S_ + vcg;

    uint4 kr0, kr1, vr0, vr1;
    kr0 = *(const uint4*)(kptr);
    kr1 = *(const uint4*)(kptr + 8);
    vr0 = *(const uint4*)(vptr);
    vr1 = *(const uint4*)(vptr + 8);

    for (int kc = 0; kc <= kmax_blk; kc += 32) {
        // --- commit staged K / V chunk to LDS ---
        *(uint4*)&Ks[keyl * KST + kcb]      = kr0;
        *(uint4*)&Ks[keyl * KST + kcb + 8]  = kr1;
        *(uint4*)&Vst[vhd * VST + vcg]      = vr0;
        *(uint4*)&Vst[vhd * VST + vcg + 8]  = vr1;
        __syncthreads();

        const int kn = kc + 32;
        if (kn <= kmax_blk) {   // prefetch next chunk
            kr0 = *(const uint4*)(kptr + (size_t)kn * H_);
            kr1 = *(const uint4*)(kptr + (size_t)kn * H_ + 8);
            vr0 = *(const uint4*)(vptr + kn);
            vr1 = *(const uint4*)(vptr + kn + 8);
        }

        if (kc <= kmax_w) {   // wave-uniform causal skip (EXEC stays all-ones)
            // --- scores[16 q][32 keys] = Q @ K^T ---
            v8f sc[2] = {vzero, vzero};
#pragma unroll
            for (int t = 0; t < 2; t++) {
#pragma unroll
                for (int c = 0; c < 4; c++) {
                    BF16Frag kb;
                    const unsigned short* p = &Ks[(t * 16 + lm) * KST + c * 32];
#pragma unroll
                    for (int v = 0; v < 4; v++)
                        kb.u[v] = *(const unsigned int*)&p[8 * lhalf + 2 * v];
#pragma unroll
                    for (int v = 0; v < 4; v++)
                        kb.u[4 + v] = *(const unsigned int*)&p[16 + 8 * lhalf + 2 * v];
                    sc[t] = wmma_bf16(qf[c].v, kb.v, sc[t]);
                }
            }
            // --- scale + rel bias + SiLU + causal mask -> bf16 staging ---
            unsigned short* abase = &Attn[wave * 16 * SST];
#pragma unroll
            for (int t = 0; t < 2; t++) {
#pragma unroll
                for (int v = 0; v < 8; v++) {
                    int M   = v + 8 * lhalf;
                    int q   = qw + M;
                    int key = kc + t * 16 + lm;
                    float x = 0.0f;
                    if (key <= q) {
                        x = sc[t][v] * scale
                          + rel_table[(size_t)(q - key + S_ - 1) * NH_ + h];
                        x = silu(x);
                    }
                    abase[M * SST + t * 16 + lm] = f2bf(x);
                }
            }
            // --- re-layout attn tile into A fragment ---
            BF16Frag afr;
            const unsigned short* ap = &Attn[(wave * 16 + lm) * SST];
#pragma unroll
            for (int v = 0; v < 4; v++)
                afr.u[v] = *(const unsigned int*)&ap[8 * lhalf + 2 * v];
#pragma unroll
            for (int v = 0; v < 4; v++)
                afr.u[4 + v] = *(const unsigned int*)&ap[16 + 8 * lhalf + 2 * v];
            // --- out += attn @ V ---
#pragma unroll
            for (int j = 0; j < 8; j++) {
                BF16Frag vb;
                const unsigned short* p = &Vst[(j * 16 + lm) * VST];
#pragma unroll
                for (int v = 0; v < 4; v++)
                    vb.u[v] = *(const unsigned int*)&p[8 * lhalf + 2 * v];
#pragma unroll
                for (int v = 0; v < 4; v++)
                    vb.u[4 + v] = *(const unsigned int*)&p[16 + 8 * lhalf + 2 * v];
                oacc[j] = wmma_bf16(afr.v, vb.v, oacc[j]);
            }
        }
        __syncthreads();
    }

    // --- epilogue: gated = out * U ---
#pragma unroll
    for (int j = 0; j < 8; j++) {
#pragma unroll
        for (int v = 0; v < 8; v++) {
            int M  = v + 8 * lhalf;
            int q  = qw + M;
            int hd = j * 16 + lm;
            size_t idx = (size_t)(b * S_ + q) * H_ + h * HD_ + hd;
            gated[idx] = f2bf(oacc[j][v] * bf2f(Up[idx]));
        }
    }
}

// ---------------------------------------------------------------------------
// Launch
// ---------------------------------------------------------------------------
extern "C" void kernel_launch(void* const* d_in, const int* in_sizes, int n_in,
                              void* d_out, int out_size, void* d_ws, size_t ws_size,
                              hipStream_t stream) {
    const float* f_query = (const float*)d_in[0];
    const float* f_key   = (const float*)d_in[1];
    const float* f_value = (const float*)d_in[2];
    // d_in[3] attn_mask: exact causal tril, computed analytically
    const float* f_Wq  = (const float*)d_in[4];
    const float* f_bq  = (const float*)d_in[5];
    const float* f_Wk  = (const float*)d_in[6];
    const float* f_bk  = (const float*)d_in[7];
    const float* f_Wv  = (const float*)d_in[8];
    const float* f_bv  = (const float*)d_in[9];
    const float* f_Wu  = (const float*)d_in[10];
    const float* f_bu  = (const float*)d_in[11];
    const float* f_Wf2 = (const float*)d_in[12];
    const float* f_bf2 = (const float*)d_in[13];
    const float* f_rel = (const float*)d_in[14];

    const size_t BSH = (size_t)B_ * S_ * H_;   // 8,388,608
    const size_t HH  = (size_t)H_ * H_;        // 4,194,304

    char* ws = (char*)d_ws;
    size_t off = 0;
    auto take = [&](size_t bytes) { char* p = ws + off; off += (bytes + 255) & ~(size_t)255; return p; };
    unsigned short* q_bf   = (unsigned short*)take(BSH * 2);
    unsigned short* k_bf   = (unsigned short*)take(BSH * 2);
    unsigned short* v_bf   = (unsigned short*)take(BSH * 2);
    unsigned short* WqT    = (unsigned short*)take(HH * 2);   // [N,K] transposed
    unsigned short* WkT    = (unsigned short*)take(HH * 2);
    unsigned short* WvT    = (unsigned short*)take(HH * 2);
    unsigned short* WuT    = (unsigned short*)take(HH * 2);
    unsigned short* Wf2T   = (unsigned short*)take(HH * 2);
    unsigned short* Qp     = (unsigned short*)take(BSH * 2);
    unsigned short* Kp     = (unsigned short*)take(BSH * 2);
    unsigned short* Vp     = (unsigned short*)take(BSH * 2);
    unsigned short* Up     = (unsigned short*)take(BSH * 2);
    unsigned short* VT     = (unsigned short*)take(BSH * 2);
    unsigned short* gated  = (unsigned short*)take(BSH * 2);

    // 1) fp32 -> bf16 conversions (activations plain, weights transposed)
    {
        int n4a = (int)(BSH / 4);
        dim3 blk(256);
        dim3 ga((n4a + 255) / 256);
        cvt_f32_to_bf16<<<ga, blk, 0, stream>>>(f_query, q_bf, n4a);
        cvt_f32_to_bf16<<<ga, blk, 0, stream>>>(f_key,   k_bf, n4a);
        cvt_f32_to_bf16<<<ga, blk, 0, stream>>>(f_value, v_bf, n4a);
        dim3 gt(H_ / 32, H_ / 32);
        cvtT_f32_to_bf16<<<gt, blk, 0, stream>>>(f_Wq,  WqT,  H_, H_);
        cvtT_f32_to_bf16<<<gt, blk, 0, stream>>>(f_Wk,  WkT,  H_, H_);
        cvtT_f32_to_bf16<<<gt, blk, 0, stream>>>(f_Wv,  WvT,  H_, H_);
        cvtT_f32_to_bf16<<<gt, blk, 0, stream>>>(f_Wu,  WuT,  H_, H_);
        cvtT_f32_to_bf16<<<gt, blk, 0, stream>>>(f_Wf2, Wf2T, H_, H_);
    }

    // 2) projections: silu(X @ W + b), bf16 out
    {
        const int M = B_ * S_, N = H_, K = H_;
        dim3 grid(N / 256, M / 128), blk(256);
        gemm_bias_act<true, false><<<grid, blk, 0, stream>>>(q_bf, WqT, f_bq, Qp, nullptr, M, N, K);
        gemm_bias_act<true, false><<<grid, blk, 0, stream>>>(k_bf, WkT, f_bk, Kp, nullptr, M, N, K);
        gemm_bias_act<true, false><<<grid, blk, 0, stream>>>(v_bf, WvT, f_bv, Vp, nullptr, M, N, K);
        gemm_bias_act<true, false><<<grid, blk, 0, stream>>>(q_bf, WuT, f_bu, Up, nullptr, M, N, K);
    }

    // 3) per-head transpose of V: [B,S,H] -> [B,NH,HD,S]
    {
        dim3 grid(S_ / 32, HD_ / 32, B_ * NH_), blk(256);
        transpose_v_bf16<<<grid, blk, 0, stream>>>(Vp, VT);
    }

    // 4) SiLU attention + U gating -> gated (bf16)
    {
        dim3 grid(S_ / 128, NH_, B_), blk(256);
        attn_silu_kernel<<<grid, blk, 0, stream>>>(Qp, Kp, VT, Up, f_rel, gated);
    }

    // 5) final GEMM: gated @ Wf2 + bf2 -> f32 out
    {
        const int M = B_ * S_, N = H_, K = H_;
        dim3 grid(N / 256, M / 128), blk(256);
        gemm_bias_act<false, true><<<grid, blk, 0, stream>>>(gated, Wf2T, f_bf2, nullptr, (float*)d_out, M, N, K);
    }
}
